// Nystroem_19481971654961
// MI455X (gfx1250) — compile-verified, matrix operand order checked by reading, not run on hardware
//
#include <hip/hip_runtime.h>

// ---------------------------------------------------------------------------
// Nystroem kernel for MI455X (gfx1250, wave32, WMMA).
//   out = exp(-gamma * (||x||^2 + ||c||^2 - 2 x c^T)) @ normalization
// Two bf16 WMMA GEMMs (v_wmma_f32_16x16x32_bf16) with f32 accumulate.
// Wave tile 64x64 (4x4 WMMA accumulators) -> 32 FLOP per cache byte, 2x the
// compute intensity of a 32x32 tile, halving L1/L2 fragment bandwidth needs.
// ---------------------------------------------------------------------------

typedef __attribute__((ext_vector_type(16))) __bf16 v16bf;
typedef __attribute__((ext_vector_type(8)))  __bf16 v8bf;
typedef __attribute__((ext_vector_type(8)))  float  v8f;

static constexpr int kN = 16384;  // rows of inputs / output
static constexpr int kC = 2048;   // centers / output cols
static constexpr int kD = 2048;   // feature dim

// ---------------------------------------------------------------------------
// Fragment loader: 16x32 bf16 tile (A-layout; B uses the same layout indexed
// by N because B's column n is a row of the (pre-transposed) operand).
//   lanes 0-15 : row = row0+lane,    K = k0 + {0..7, 16..23}
//   lanes 16-31: row = row0+lane-16, K = k0 + {8..15, 24..31}
// Each half is 8 contiguous bf16 = one global_load_b128.
// ---------------------------------------------------------------------------
__device__ __forceinline__ v16bf load_frag(const __bf16* __restrict__ base,
                                           int row0, int k0, int ld, int lane) {
  const int r  = row0 + (lane & 15);
  const int kk = k0 + ((lane >> 4) << 3);
  const __bf16* p = base + (size_t)r * ld + kk;
  v8bf lo = *reinterpret_cast<const v8bf*>(p);
  v8bf hi = *reinterpret_cast<const v8bf*>(p + 16);
  v16bf f;
#pragma unroll
  for (int i = 0; i < 8; ++i) { f[i] = lo[i]; f[i + 8] = hi[i]; }
  return f;
}

// ---------------------------------------------------------------------------
// f32 -> bf16 row conversion fused with row ||.||^2 (one block per row).
// ---------------------------------------------------------------------------
__global__ void __launch_bounds__(256)
convert_rowsq_kernel(const float* __restrict__ src, __bf16* __restrict__ dst,
                     float* __restrict__ rowsq, int D) {
  const int row = blockIdx.x;
  const float* p = src + (size_t)row * D;
  __bf16* q = dst + (size_t)row * D;
  float s = 0.0f;
  for (int i = threadIdx.x; i < D; i += 256) {
    float v = p[i];
    s += v * v;
    q[i] = (__bf16)v;
  }
  __shared__ float red[256];
  red[threadIdx.x] = s;
  __syncthreads();
#pragma unroll
  for (int off = 128; off > 0; off >>= 1) {
    if (threadIdx.x < off) red[threadIdx.x] += red[threadIdx.x + off];
    __syncthreads();
  }
  if (threadIdx.x == 0) rowsq[row] = red[0];
}

// ---------------------------------------------------------------------------
// Transpose + convert: dst[j][k] = (bf16)src[k][j].  LDS-tiled 32x32 so both
// the f32 reads and bf16 writes are contiguous.
// ---------------------------------------------------------------------------
__global__ void __launch_bounds__(256)
transpose_convert_kernel(const float* __restrict__ src, __bf16* __restrict__ dst,
                         int C) {
  __shared__ float tile[32][33];
  const int bx = blockIdx.x * 32;  // src column block
  const int by = blockIdx.y * 32;  // src row block
  const int tx = threadIdx.x & 31;
  const int ty = threadIdx.x >> 5;  // 0..7
#pragma unroll
  for (int r = ty; r < 32; r += 8)
    tile[r][tx] = src[(size_t)(by + r) * C + (bx + tx)];
  __syncthreads();
#pragma unroll
  for (int r = ty; r < 32; r += 8)
    dst[(size_t)(bx + r) * C + (by + tx)] = (__bf16)tile[tx][r];
}

// ---------------------------------------------------------------------------
// Shared GEMM core: 8 waves as 2(M) x 4(N); block tile 128x256.
// Wave tile 64x64 = 4x4 WMMA f32 accumulators; K loop steps of 32.
// ---------------------------------------------------------------------------
struct WaveTile {
  v8f acc[4][4];
  int m0, n0, lane;
};

__device__ __forceinline__ void gemm_core(WaveTile& t,
                                          const __bf16* __restrict__ A,
                                          const __bf16* __restrict__ B,
                                          int K) {
  const int lane = threadIdx.x & 31;
  const int w    = threadIdx.x >> 5;
  const int wm   = w & 1;    // 0..1
  const int wn   = w >> 1;   // 0..3
  t.m0   = blockIdx.x * 128 + wm * 64;
  t.n0   = blockIdx.y * 256 + wn * 64;
  t.lane = lane;
#pragma unroll
  for (int mi = 0; mi < 4; ++mi)
#pragma unroll
    for (int ni = 0; ni < 4; ++ni) t.acc[mi][ni] = (v8f){};

  for (int k0 = 0; k0 < K; k0 += 32) {
    v16bf a[4], b[4];
#pragma unroll
    for (int mi = 0; mi < 4; ++mi) a[mi] = load_frag(A, t.m0 + mi * 16, k0, K, lane);
#pragma unroll
    for (int ni = 0; ni < 4; ++ni) b[ni] = load_frag(B, t.n0 + ni * 16, k0, K, lane);
#pragma unroll
    for (int mi = 0; mi < 4; ++mi)
#pragma unroll
      for (int ni = 0; ni < 4; ++ni)
        t.acc[mi][ni] = __builtin_amdgcn_wmma_f32_16x16x32_bf16(
            false, a[mi], false, b[ni], (short)0, t.acc[mi][ni], false, false);
  }
}

// ---------------------------------------------------------------------------
// GEMM1 + Gaussian epilogue:
//   dens[i][j] = (bf16) exp(-g*(x_sq[i] + c_sq[j] - 2 * <inputs_i, centers_j>))
// ---------------------------------------------------------------------------
__global__ void __launch_bounds__(256)
gemm1_density_kernel(const __bf16* __restrict__ Abf,   // [N,D] inputs bf16
                     const __bf16* __restrict__ Bbf,   // [C,D] centers bf16
                     const float* __restrict__ x_sq,   // [N]
                     const float* __restrict__ c_sq,   // [C]
                     const float* __restrict__ gamma,  // [1]
                     __bf16* __restrict__ dens,        // [N,C]
                     int Cc, int Dd) {
  WaveTile t;
  gemm_core(t, Abf, Bbf, Dd);

  const float g  = gamma[0];
  const int   ln = t.lane & 15;
  const int   lh = t.lane >> 4;
#pragma unroll
  for (int mi = 0; mi < 4; ++mi) {
#pragma unroll
    for (int ni = 0; ni < 4; ++ni) {
      const int col = t.n0 + ni * 16 + ln;
      const float cs = c_sq[col];
#pragma unroll
      for (int v = 0; v < 8; ++v) {
        const int row = t.m0 + mi * 16 + lh * 8 + v;
        const float sq = x_sq[row] + cs - 2.0f * t.acc[mi][ni][v];
        dens[(size_t)row * Cc + col] = (__bf16)__expf(-g * sq);
      }
    }
  }
}

// ---------------------------------------------------------------------------
// GEMM2: out = dens(bf16) @ normalization (via pre-transposed bf16 norm^T).
// ---------------------------------------------------------------------------
__global__ void __launch_bounds__(256)
gemm2_kernel(const __bf16* __restrict__ Abf,    // [N,C] densities bf16
             const __bf16* __restrict__ BTbf,   // [C,C] norm^T bf16
             float* __restrict__ out,           // [N,C]
             int Cc) {
  WaveTile t;
  gemm_core(t, Abf, BTbf, Cc);

  const int ln = t.lane & 15;
  const int lh = t.lane >> 4;
#pragma unroll
  for (int mi = 0; mi < 4; ++mi) {
#pragma unroll
    for (int ni = 0; ni < 4; ++ni) {
      const int col = t.n0 + ni * 16 + ln;
#pragma unroll
      for (int v = 0; v < 8; ++v) {
        const int row = t.m0 + mi * 16 + lh * 8 + v;
        out[(size_t)row * Cc + col] = t.acc[mi][ni][v];
      }
    }
  }
}

// ---------------------------------------------------------------------------
// Launch
// ---------------------------------------------------------------------------
extern "C" void kernel_launch(void* const* d_in, const int* in_sizes, int n_in,
                              void* d_out, int out_size, void* d_ws, size_t ws_size,
                              hipStream_t stream) {
  const float* inputs  = (const float*)d_in[0];  // [N,D]
  const float* centers = (const float*)d_in[1];  // [C,D]
  const float* gamma   = (const float*)d_in[2];  // [1]
  const float* norm    = (const float*)d_in[3];  // [C,C]
  float* out = (float*)d_out;                    // [N,C]

  // Workspace carve-up (256B aligned). Total ~144.1 MB.
  char* ws = (char*)d_ws;
  size_t off = 0;
  auto carve = [&](size_t bytes) -> void* {
    off = (off + 255) & ~(size_t)255;
    void* p = ws + off;
    off += bytes;
    return p;
  };
  __bf16* inputs_bf  = (__bf16*)carve((size_t)kN * kD * sizeof(__bf16)); // 64 MB
  __bf16* centers_bf = (__bf16*)carve((size_t)kC * kD * sizeof(__bf16)); //  8 MB
  __bf16* normT_bf   = (__bf16*)carve((size_t)kC * kC * sizeof(__bf16)); //  8 MB
  __bf16* dens_bf    = (__bf16*)carve((size_t)kN * kC * sizeof(__bf16)); // 64 MB
  float*  x_sq       = (float*) carve((size_t)kN * sizeof(float));
  float*  c_sq       = (float*) carve((size_t)kC * sizeof(float));
  (void)ws_size;

  // Pre-pass: bf16 conversions + row squared-norms + transposed normalization.
  convert_rowsq_kernel<<<dim3(kN), dim3(256), 0, stream>>>(inputs, inputs_bf, x_sq, kD);
  convert_rowsq_kernel<<<dim3(kC), dim3(256), 0, stream>>>(centers, centers_bf, c_sq, kD);
  transpose_convert_kernel<<<dim3(kC / 32, kC / 32), dim3(256), 0, stream>>>(norm, normT_bf, kC);

  // GEMM1 (inputs @ centers^T) fused with Gaussian epilogue -> densities bf16.
  gemm1_density_kernel<<<dim3(kN / 128, kC / 256), dim3(256), 0, stream>>>(
      inputs_bf, centers_bf, x_sq, c_sq, gamma, dens_bf, kC, kD);

  // GEMM2 (densities @ normalization) -> f32 output.
  gemm2_kernel<<<dim3(kN / 128, kC / 256), dim3(256), 0, stream>>>(
      dens_bf, normT_bf, out, kC);
}
